// BatchHardTripletLoss_63582695850911
// MI455X (gfx1250) — matrix operand order repaired and checked by reading, not compiled, not run on hardware
//
#include <hip/hip_runtime.h>
#include <math.h>
#include <stdint.h>

typedef __attribute__((ext_vector_type(16))) _Float16 v16h;
typedef __attribute__((ext_vector_type(8)))  _Float16 v8h;
typedef __attribute__((ext_vector_type(8)))  float    v8f;
typedef __attribute__((ext_vector_type(4)))  int      v4i;

#define BIGV 1.0e9f

// ---------------------------------------------------------------------------
// Kernel 1: L2-normalize rows (fp32 in) -> fp16 normalized matrix in ws
// ---------------------------------------------------------------------------
__global__ void bhtl_normalize_kernel(const float* __restrict__ X,
                                      _Float16* __restrict__ Xn,
                                      int N, int D) {
    const int row = blockIdx.x;
    const float* xr = X + (size_t)row * D;
    float ssum = 0.f;
    for (int c = threadIdx.x; c < D; c += blockDim.x) {
        float v = xr[c];
        ssum += v * v;
    }
    #pragma unroll
    for (int off = 16; off >= 1; off >>= 1)
        ssum += __shfl_xor(ssum, off);

    __shared__ float wsum[8];
    const int lane = threadIdx.x & 31;
    const int wid  = threadIdx.x >> 5;
    if (lane == 0) wsum[wid] = ssum;
    __syncthreads();
    const int nw = blockDim.x >> 5;
    float tot = 0.f;
    for (int w = 0; w < nw; ++w) tot += wsum[w];

    const float inv = 1.0f / fmaxf(sqrtf(tot), 1e-12f);
    for (int c = threadIdx.x; c < D; c += blockDim.x)
        Xn[(size_t)row * D + c] = (_Float16)(xr[c] * inv);
}

// ---------------------------------------------------------------------------
// Masked per-tile statistics update (epilogue of one 16x16 C tile).
// ---------------------------------------------------------------------------
template <int PASS>
__device__ __forceinline__ void tile_epilogue(
    const v8f& c, int col, int i0, int hiRow,
    const int* __restrict__ pidR, const float* __restrict__ hpR,
    const int* __restrict__ labels, const int* __restrict__ pids,
    float* acc0, float* acc1, float* acc2, float* acc3) {
    const int  pidC = pids[col];
    const bool genC = (labels[col] == 1);
    #pragma unroll
    for (int r = 0; r < 8; ++r) {
        const int   row  = i0 + hiRow + r;
        const float s    = c[r];
        const bool  same = (pidR[r] == pidC);
        const bool  pos  = same && genC && (row != col);
        const bool  neg  = (!same) || (!genC);
        if (PASS == 1) {
            acc0[r] = pos ? fminf(acc0[r], s) : acc0[r];
            acc1[r] = neg ? fmaxf(acc1[r], s) : acc1[r];
            acc2[r] = pos ? 1.f : acc2[r];
            acc3[r] = neg ? 1.f : acc3[r];
        } else {
            const bool semi = neg && (s < hpR[r]) && (s > hpR[r] - 0.5f);
            const bool cand = neg && (s > hpR[r]);
            acc0[r] = semi ? fmaxf(acc0[r], s) : acc0[r];
            acc1[r] = cand ? fmaxf(acc1[r], s) : acc1[r];
            acc2[r] = semi ? 1.f : acc2[r];
            acc3[r] = cand ? 1.f : acc3[r];
        }
    }
}

// ---------------------------------------------------------------------------
// Kernels 2/3: WMMA similarity sweeps.
//   PASS 1: hp = min over positives, hn_all = max over negatives, any-flags
//   PASS 2: hn_semi / hn_cand given hp; emits per-row loss*valid and valid
// DK > 0 : compile-time K. The block's 16-row A strip is staged in LDS once
//          (padded rows -> conflict-minimal b128 access) and shared by all
//          8 waves. A fragments are fetched with explicit ds_load_b128
//          inline asm (pinned in-loop, DScnt-tracked, one s_wait_dscnt per
//          K chunk) while B streams via global loads on LOADcnt — the two
//          counters overlap. Two column tiles per iteration give two
//          independent WMMA chains.
// DK == 0: generic runtime-K fallback, single tile per iteration.
// ---------------------------------------------------------------------------
template <int PASS, int DK>
__launch_bounds__(256)
__global__ void bhtl_pass_kernel(const _Float16* __restrict__ Xn,
                                 const int* __restrict__ labels,
                                 const int* __restrict__ pids,
                                 float* __restrict__ hp_out,
                                 float* __restrict__ hnall_out,
                                 float* __restrict__ pa_out,
                                 float* __restrict__ na_out,
                                 const float* __restrict__ hp_in,
                                 const float* __restrict__ hnall_in,
                                 const float* __restrict__ pa_in,
                                 const float* __restrict__ na_in,
                                 float* __restrict__ lossv,
                                 float* __restrict__ vcnt,
                                 int N, int D) {
    const int i0     = blockIdx.x << 4;       // 16-row strip
    const int lane   = threadIdx.x & 31;
    const int wave   = threadIdx.x >> 5;
    const int nwaves = blockDim.x >> 5;       // 8
    const int half16 = lane >> 4;             // 0: lanes 0-15, 1: lanes 16-31
    const int l15    = lane & 15;
    const int hiRow  = half16 << 3;           // rows r / 8+r per C-layout

    int   pidR[8];
    float hpR[8];
    #pragma unroll
    for (int r = 0; r < 8; ++r) {
        const int row = i0 + hiRow + r;
        pidR[r] = pids[row];
        hpR[r]  = (PASS == 2) ? hp_in[row] : 0.f;
    }

    float acc0[8], acc1[8], acc2[8], acc3[8];
    #pragma unroll
    for (int r = 0; r < 8; ++r) {
        acc0[r] = (PASS == 1) ? BIGV : -BIGV;
        acc1[r] = -BIGV;
        acc2[r] = 0.f;
        acc3[r] = 0.f;
    }

    // A 16x32 layout (contiguous, matching B): lanes 0-15 hold K 0..15,
    // lanes 16-31 hold K 16..31 of their row.
    const int aoff = half16 << 4;    // 0 or 16 halfs
    const int boff = half16 << 4;    // B: lanes 0-15 K 0..15, lanes 16-31 K 16..31

    if (DK > 0) {
        constexpr int KD    = (DK > 0 ? DK : 32);
        constexpr int KC    = KD / 32;          // K chunks of 32
        constexpr int APAD  = 8;                // halfs: row stride 132 DWORDs
        constexpr int ASTR  = KD + APAD;

        // Stage the block's A strip (16 rows x KD halfs) into LDS once.
        __shared__ _Float16 As[16 * ASTR];
        {
            constexpr int CH = KD / 8;          // 16-byte chunks per row
            for (int idx = threadIdx.x; idx < 16 * CH; idx += blockDim.x) {
                const int row = idx / CH;
                const int ch  = idx - row * CH;
                *(v8h*)(&As[row * ASTR + ch * 8]) =
                    *(const v8h*)(Xn + (size_t)(i0 + row) * KD + ch * 8);
            }
        }
        __syncthreads();

        // LDS byte address of this lane's A row + K-phase offset. Generic
        // pointers to LDS carry the DS byte offset in their low 32 bits.
        const unsigned aBase =
            (unsigned)(uintptr_t)(&As[l15 * ASTR + aoff]);

        // Two column tiles (32 columns) per iteration: two independent
        // WMMA accumulation chains sharing each A fragment.
        const int nblk = N >> 5;
        for (int t = wave; t < nblk; t += nwaves) {
            const int j0 = t << 5;
            const _Float16* B0 = Xn + (size_t)(j0 + l15) * KD;
            const _Float16* B1 = Xn + (size_t)(j0 + 16 + l15) * KD;

            v8f c0 = {0.f, 0.f, 0.f, 0.f, 0.f, 0.f, 0.f, 0.f};
            v8f c1 = {0.f, 0.f, 0.f, 0.f, 0.f, 0.f, 0.f, 0.f};
            #pragma unroll
            for (int k = 0; k < KC; ++k) {
                // A fragment: two 16B halves via explicit DS loads, one
                // combined DScnt wait. Early-clobber so destinations never
                // alias the address VGPR.
                const unsigned a0addr = aBase + (unsigned)(k * 64);
                v4i dlo, dhi;
                asm volatile(
                    "ds_load_b128 %0, %2\n\t"
                    "ds_load_b128 %1, %2 offset:16\n\t"
                    "s_wait_dscnt 0x0"
                    : "=&v"(dlo), "=&v"(dhi)
                    : "v"(a0addr));
                const v8h alo = __builtin_bit_cast(v8h, dlo);
                const v8h ahi = __builtin_bit_cast(v8h, dhi);
                v16h a;
                #pragma unroll
                for (int u = 0; u < 8; ++u) { a[u] = alo[u]; a[u + 8] = ahi[u]; }

                v16h b0 = *(const v16h*)(B0 + k * 32 + boff);
                v16h b1 = *(const v16h*)(B1 + k * 32 + boff);
                c0 = __builtin_amdgcn_wmma_f32_16x16x32_f16(
                        false, a, false, b0, (short)0, c0, false, false);
                c1 = __builtin_amdgcn_wmma_f32_16x16x32_f16(
                        false, a, false, b1, (short)0, c1, false, false);
            }
            tile_epilogue<PASS>(c0, j0 + l15,      i0, hiRow, pidR, hpR,
                                labels, pids, acc0, acc1, acc2, acc3);
            tile_epilogue<PASS>(c1, j0 + 16 + l15, i0, hiRow, pidR, hpR,
                                labels, pids, acc0, acc1, acc2, acc3);
        }
    } else {
        // Generic runtime-K fallback (single tile per iteration).
        const _Float16* Arow = Xn + (size_t)(i0 + l15) * D;
        const int ntiles = N >> 4;
        for (int t = wave; t < ntiles; t += nwaves) {
            const int j0 = t << 4;
            const _Float16* Brow = Xn + (size_t)(j0 + l15) * D;
            v8f c = {0.f, 0.f, 0.f, 0.f, 0.f, 0.f, 0.f, 0.f};
            for (int kc = 0; kc < D; kc += 32) {
                v8h alo = *(const v8h*)(Arow + kc + aoff);
                v8h ahi = *(const v8h*)(Arow + kc + 8 + aoff);
                v16h a;
                #pragma unroll
                for (int u = 0; u < 8; ++u) { a[u] = alo[u]; a[u + 8] = ahi[u]; }
                v16h b = *(const v16h*)(Brow + kc + boff);
                c = __builtin_amdgcn_wmma_f32_16x16x32_f16(
                        false, a, false, b, (short)0, c, false, false);
            }
            tile_epilogue<PASS>(c, j0 + l15, i0, hiRow, pidR, hpR,
                                labels, pids, acc0, acc1, acc2, acc3);
        }
    }

    // Cross-lane reduce within each 16-lane half (xor<16 stays in half)
    #pragma unroll
    for (int off = 8; off >= 1; off >>= 1) {
        #pragma unroll
        for (int r = 0; r < 8; ++r) {
            const float o0 = __shfl_xor(acc0[r], off);
            acc0[r] = (PASS == 1) ? fminf(acc0[r], o0) : fmaxf(acc0[r], o0);
            acc1[r] = fmaxf(acc1[r], __shfl_xor(acc1[r], off));
            acc2[r] = fmaxf(acc2[r], __shfl_xor(acc2[r], off));
            acc3[r] = fmaxf(acc3[r], __shfl_xor(acc3[r], off));
        }
    }

    // Combine the 8 waves through LDS
    __shared__ float smem[4][8][16];   // [stat][wave][rowInStrip]
    if (l15 == 0) {
        #pragma unroll
        for (int r = 0; r < 8; ++r) {
            smem[0][wave][hiRow + r] = acc0[r];
            smem[1][wave][hiRow + r] = acc1[r];
            smem[2][wave][hiRow + r] = acc2[r];
            smem[3][wave][hiRow + r] = acc3[r];
        }
    }
    __syncthreads();

    if (threadIdx.x < 16) {
        const int ri  = threadIdx.x;
        const int row = i0 + ri;
        float a0 = smem[0][0][ri], a1 = smem[1][0][ri];
        float a2 = smem[2][0][ri], a3 = smem[3][0][ri];
        for (int w = 1; w < nwaves; ++w) {
            a0 = (PASS == 1) ? fminf(a0, smem[0][w][ri]) : fmaxf(a0, smem[0][w][ri]);
            a1 = fmaxf(a1, smem[1][w][ri]);
            a2 = fmaxf(a2, smem[2][w][ri]);
            a3 = fmaxf(a3, smem[3][w][ri]);
        }
        if (PASS == 1) {
            hp_out[row]    = a0;
            hnall_out[row] = a1;
            pa_out[row]    = a2;
            na_out[row]    = a3;
        } else {
            const float hp = hp_in[row];
            const float hn = (a2 > 0.f) ? a0 : ((a3 > 0.f) ? a1 : hnall_in[row]);
            const bool valid = (labels[row] == 1) && (pa_in[row] > 0.f) && (na_in[row] > 0.f);
            const float base = fmaxf(hn - hp + 0.5f, 0.f);
            const float wgt  = ((hp < 0.6f) || (hn > 0.3f)) ? 2.0f : 1.0f;
            const float loss = base * wgt + 0.5f * (1.0f - hp)
                             + 0.5f * fmaxf(hn + 0.2f, 0.f);
            lossv[row] = valid ? loss : 0.f;
            vcnt[row]  = valid ? 1.f : 0.f;
        }
    }
}

// ---------------------------------------------------------------------------
// Kernel 4: masked mean -> scalar
// ---------------------------------------------------------------------------
__global__ void bhtl_finalize_kernel(const float* __restrict__ lossv,
                                     const float* __restrict__ vcnt,
                                     float* __restrict__ out, int N) {
    float ts = 0.f, tc = 0.f;
    for (int i = threadIdx.x; i < N; i += blockDim.x) {
        ts += lossv[i];
        tc += vcnt[i];
    }
    #pragma unroll
    for (int off = 16; off >= 1; off >>= 1) {
        ts += __shfl_xor(ts, off);
        tc += __shfl_xor(tc, off);
    }
    __shared__ float ss[8], sc[8];
    const int lane = threadIdx.x & 31;
    const int wid  = threadIdx.x >> 5;
    if (lane == 0) { ss[wid] = ts; sc[wid] = tc; }
    __syncthreads();
    if (threadIdx.x == 0) {
        float S = 0.f, C = 0.f;
        const int nw = blockDim.x >> 5;
        for (int w = 0; w < nw; ++w) { S += ss[w]; C += sc[w]; }
        out[0] = (C > 0.f) ? (S / C) : 0.f;
    }
}

// ---------------------------------------------------------------------------
extern "C" void kernel_launch(void* const* d_in, const int* in_sizes, int n_in,
                              void* d_out, int out_size, void* d_ws, size_t ws_size,
                              hipStream_t stream) {
    const float* emb    = (const float*)d_in[0];
    const int*   labels = (const int*)d_in[1];
    const int*   pids   = (const int*)d_in[2];

    const int N = in_sizes[1];                 // labels: (N,)
    const int D = in_sizes[0] / N;             // embeddings: (N, D)

    char* ws = (char*)d_ws;
    _Float16* Xn = (_Float16*)ws;
    size_t off = (size_t)N * (size_t)D * sizeof(_Float16);
    float* hp    = (float*)(ws + off);
    float* hnall = hp + N;
    float* pa    = hnall + N;
    float* na    = pa + N;
    float* lossv = na + N;
    float* vcnt  = lossv + N;

    bhtl_normalize_kernel<<<N, 256, 0, stream>>>(emb, Xn, N, D);

    const int nstrips = N >> 4;
    if (D == 256 && (N & 31) == 0) {
        bhtl_pass_kernel<1, 256><<<nstrips, 256, 0, stream>>>(
            Xn, labels, pids, hp, hnall, pa, na,
            hp, hnall, pa, na, lossv, vcnt, N, D);
        bhtl_pass_kernel<2, 256><<<nstrips, 256, 0, stream>>>(
            Xn, labels, pids, hp, hnall, pa, na,
            hp, hnall, pa, na, lossv, vcnt, N, D);
    } else {
        bhtl_pass_kernel<1, 0><<<nstrips, 256, 0, stream>>>(
            Xn, labels, pids, hp, hnall, pa, na,
            hp, hnall, pa, na, lossv, vcnt, N, D);
        bhtl_pass_kernel<2, 0><<<nstrips, 256, 0, stream>>>(
            Xn, labels, pids, hp, hnall, pa, na,
            hp, hnall, pa, na, lossv, vcnt, N, D);
    }

    bhtl_finalize_kernel<<<1, 256, 0, stream>>>(lossv, vcnt, (float*)d_out, N);
}
